// ScaledAttention_1005022347679
// MI455X (gfx1250) — matrix operand order repaired and checked by reading, not compiled, not run on hardware
//
#include <hip/hip_runtime.h>

typedef __attribute__((ext_vector_type(16))) _Float16 v16h;
typedef __attribute__((ext_vector_type(8)))  _Float16 v8h;
typedef __attribute__((ext_vector_type(8)))  float    v8f;

#define HEAD   64
#define EMB    1024
#define TSEQ   2048
#define NBATCH 8
#define NEGBIG (-1.0e30f)

static __device__ __forceinline__ v8f wmma_f16(v16h a, v16h b, v8f c) {
  // D = A(16x32 f16) * B(32x16 f16) + C(16x16 f32)
  return __builtin_amdgcn_wmma_f32_16x16x32_f16(false, a, false, b, (short)0, c, false, false);
}

// ---------------------------------------------------------------------------
// Kernel 0: one-time fp32 -> f16 conversion of [Wq; Wk; Wv] -> Wh (192, 1024).
// ---------------------------------------------------------------------------
__global__ __launch_bounds__(256)
void convert_w_kernel(const float* __restrict__ Wq,
                      const float* __restrict__ Wk,
                      const float* __restrict__ Wv,
                      _Float16* __restrict__ Wh)
{
  const int idx = (blockIdx.x * 256 + threadIdx.x) * 4;   // 4 elems / thread
  const int per = HEAD * EMB;                             // 65536
  const float* src = (idx < per) ? Wq : (idx < 2 * per) ? Wk : Wv;
  const int    off = idx % per;
  float4 w = *(const float4*)(src + off);
  _Float16* dst = Wh + idx;
  dst[0] = (_Float16)w.x; dst[1] = (_Float16)w.y;
  dst[2] = (_Float16)w.z; dst[3] = (_Float16)w.w;
}

// ---------------------------------------------------------------------------
// Kernel 1: Q = (X Wq^T) * C^-0.5, K = X Wk^T  (row-major (B*T,H) f16)
//           Vt = (X Wv^T)^T                    ((B,H,T) f16, for contiguous B-frags)
// One wave per 16-row tile of X; K-loop over C in steps of 32.
// An empty asm fence pins all 13 fragments live before the WMMA chain so the
// loads issue as one clause and overlap the 12 WMMAs, instead of the
// load->wait0->wmma serialization the scheduler produces otherwise.
// ---------------------------------------------------------------------------
__global__ __launch_bounds__(256)
void qkv_proj_kernel(const float* __restrict__ X,
                     const _Float16* __restrict__ Wh,   // (192, 1024) f16
                     _Float16* __restrict__ Qh,
                     _Float16* __restrict__ Kh,
                     _Float16* __restrict__ Vt)
{
  const int lane = threadIdx.x & 31;
  const int wave = threadIdx.x >> 5;
  const int l16  = lane & 15;
  const int half = lane >> 4;

  const int tile = blockIdx.x * 8 + wave;   // 1024 tiles of 16 rows over B*T
  const int m0   = tile << 4;

  const v8f vzero = {0.f,0.f,0.f,0.f,0.f,0.f,0.f,0.f};
  v8f acc[12];
#pragma unroll
  for (int i = 0; i < 12; ++i) acc[i] = vzero;

  const size_t arow_off = (size_t)(m0 + l16) * EMB;
  const _Float16* wrow = Wh + (size_t)l16 * EMB;

  for (int kt = 0; kt < EMB / 32; ++kt) {
    const int k0 = kt * 32;
    const int bk = k0 + half * 16;  // B: lanes 0-15 hold K 0..15, lanes 16-31 hold K 16..31

    // Issue all 12 B-fragment loads first.
    v16h bfr[12];
#pragma unroll
    for (int c = 0; c < 12; ++c)    // 12 output col tiles: Q(0-3) K(4-7) V(8-11)
      bfr[c] = *(const v16h*)(wrow + (size_t)(c * 16) * EMB + bk);

    // A fragment: lane(0-15)=row, K = half*16 + {0..15}
    v16h a;
    {
      const float4* xp = (const float4*)(X + arow_off + k0 + half * 16);
      float4 x0 = xp[0], x1 = xp[1], x2 = xp[2], x3 = xp[3];
      a[0]=(_Float16)x0.x;  a[1]=(_Float16)x0.y;  a[2]=(_Float16)x0.z;  a[3]=(_Float16)x0.w;
      a[4]=(_Float16)x1.x;  a[5]=(_Float16)x1.y;  a[6]=(_Float16)x1.z;  a[7]=(_Float16)x1.w;
      a[8]=(_Float16)x2.x;  a[9]=(_Float16)x2.y;  a[10]=(_Float16)x2.z; a[11]=(_Float16)x2.w;
      a[12]=(_Float16)x3.x; a[13]=(_Float16)x3.y; a[14]=(_Float16)x3.z; a[15]=(_Float16)x3.w;
    }

    // Materialization fence: force every fragment resident before any WMMA.
    asm volatile("" ::
      "v"(bfr[0]), "v"(bfr[1]), "v"(bfr[2]),  "v"(bfr[3]),
      "v"(bfr[4]), "v"(bfr[5]), "v"(bfr[6]),  "v"(bfr[7]),
      "v"(bfr[8]), "v"(bfr[9]), "v"(bfr[10]), "v"(bfr[11]),
      "v"(a));

#pragma unroll
    for (int c = 0; c < 12; ++c)
      acc[c] = wmma_f16(a, bfr[c], acc[c]);
  }

  // Epilogue. D layout: vgpr i, lanes 0-15 -> row i; lanes 16-31 -> row i+8; col = l16.
  const int   b  = m0 / TSEQ;
  const int   t0 = m0 % TSEQ;
  const float qs = 0.03125f;   // EMB^-0.5 = 1/32 (exact)
#pragma unroll
  for (int ct = 0; ct < 4; ++ct) {
    const int h = ct * 16 + l16;
#pragma unroll
    for (int i = 0; i < 8; ++i) {
      const size_t row = (size_t)(m0 + half * 8 + i);
      Qh[row * HEAD + h] = (_Float16)(acc[ct][i] * qs);
      Kh[row * HEAD + h] = (_Float16)(acc[4 + ct][i]);
    }
    // V transposed: Vt[b][h][t]; fixed lane -> fixed h, i varies t -> contiguous store
    v8h vv;
#pragma unroll
    for (int i = 0; i < 8; ++i) vv[i] = (_Float16)(acc[8 + ct][i]);
    *(v8h*)(Vt + ((size_t)b * HEAD + h) * TSEQ + t0 + half * 8) = vv;
  }
}

// ---------------------------------------------------------------------------
// Kernel 2: flash attention, one 16-query tile per wave, 32 keys per step.
// K and V fragments both load at the top of each iteration (fenced), so
// V-load latency hides under score WMMAs + softmax VALU + LDS transpose.
// ---------------------------------------------------------------------------
__global__ __launch_bounds__(256)
void flash_attn_kernel(const _Float16* __restrict__ Qh,
                       const _Float16* __restrict__ Kh,
                       const _Float16* __restrict__ Vt,
                       const int* __restrict__ pmask,
                       float* __restrict__ out)
{
  __shared__ float pbuf[8][16 * 36];   // per-wave P scratch, padded stride
  const int lane = threadIdx.x & 31;
  const int wave = threadIdx.x >> 5;
  const int l16  = lane & 15;
  const int half = lane >> 4;

  const int qt  = blockIdx.x * 8 + wave;  // 1024 query tiles
  const int gq0 = qt << 4;                // global row base in (B*T)
  const int b   = gq0 / TSEQ;

  // Q A-fragments (held all kernel): row = l16, K = half*16 + {0..15} (+32 for aq1)
  v16h aq0, aq1;
  {
    const _Float16* qp = Qh + (size_t)(gq0 + l16) * HEAD + half * 16;
    v8h q0 = *(const v8h*)(qp);
    v8h q1 = *(const v8h*)(qp + 8);
    v8h q2 = *(const v8h*)(qp + 32);
    v8h q3 = *(const v8h*)(qp + 40);
#pragma unroll
    for (int e = 0; e < 8; ++e) {
      aq0[e] = q0[e]; aq0[8 + e] = q1[e];
      aq1[e] = q2[e]; aq1[8 + e] = q3[e];
    }
  }

  const v8f vzero = {0.f,0.f,0.f,0.f,0.f,0.f,0.f,0.f};
  v8f accf[4];
#pragma unroll
  for (int f = 0; f < 4; ++f) accf[f] = vzero;
  float mrow[8], lrow[8];
#pragma unroll
  for (int i = 0; i < 8; ++i) { mrow[i] = NEGBIG; lrow[i] = 0.f; }

  const int*      mk    = pmask + b * TSEQ;
  float*          pb    = pbuf[wave];
  const _Float16* kbase = Kh + (size_t)b * TSEQ * HEAD;
  const _Float16* vbase = Vt + (size_t)b * HEAD * TSEQ;

  for (int s0 = 0; s0 < TSEQ; s0 += 32) {
    // ---- issue ALL global loads for this iteration up front ----
    // K B-frags: col n = key s0(+16)+l16, 16 contiguous h values at half*16 (+32)
    const _Float16* kp0 = kbase + (size_t)(s0 + l16) * HEAD + half * 16;
    v16h bk00 = *(const v16h*)(kp0);
    v16h bk01 = *(const v16h*)(kp0 + 32);
    v16h bk10 = *(const v16h*)(kp0 + 16 * HEAD);
    v16h bk11 = *(const v16h*)(kp0 + 16 * HEAD + 32);
    // V B-frags: B[k][n] = V[s0+k][h] = Vt[h][s0+k], contiguous per lane
    v16h bv[4];
#pragma unroll
    for (int f = 0; f < 4; ++f)
      bv[f] = *(const v16h*)(vbase + (size_t)(f * 16 + l16) * TSEQ + s0 + half * 16);
    const bool km0 = mk[s0 + l16] != 0;
    const bool km1 = mk[s0 + 16 + l16] != 0;

    // Materialization fence: V frags resident before the score WMMAs begin.
    asm volatile("" ::
      "v"(bk00), "v"(bk01), "v"(bk10), "v"(bk11),
      "v"(bv[0]), "v"(bv[1]), "v"(bv[2]), "v"(bv[3]));

    // ---- scores ----
    v8f S0 = wmma_f16(aq0, bk00, vzero);
    S0     = wmma_f16(aq1, bk01, S0);
    v8f S1 = wmma_f16(aq0, bk10, vzero);
    S1     = wmma_f16(aq1, bk11, S1);

    float t0[8], t1[8], red[8];
#pragma unroll
    for (int i = 0; i < 8; ++i) {
      t0[i]  = km0 ? NEGBIG : S0[i];
      t1[i]  = km1 ? NEGBIG : S1[i];
      red[i] = fmaxf(t0[i], t1[i]);
    }
    // row max across 16 lanes (xor 1/2/4/8 stays within each 16-lane half)
#pragma unroll
    for (int d = 1; d <= 8; d <<= 1)
#pragma unroll
      for (int i = 0; i < 8; ++i)
        red[i] = fmaxf(red[i], __shfl_xor(red[i], d, 32));

    float esc[8];
#pragma unroll
    for (int i = 0; i < 8; ++i) {
      const float mn = fmaxf(mrow[i], red[i]);
      esc[i]  = __expf(mrow[i] - mn);
      mrow[i] = mn;
      t0[i]   = __expf(t0[i] - mn);
      t1[i]   = __expf(t1[i] - mn);
      red[i]  = t0[i] + t1[i];
    }
#pragma unroll
    for (int d = 1; d <= 8; d <<= 1)
#pragma unroll
      for (int i = 0; i < 8; ++i)
        red[i] += __shfl_xor(red[i], d, 32);
#pragma unroll
    for (int i = 0; i < 8; ++i) lrow[i] = lrow[i] * esc[i] + red[i];
#pragma unroll
    for (int f = 0; f < 4; ++f)
#pragma unroll
      for (int i = 0; i < 8; ++i) accf[f][i] *= esc[i];

    // P (D-layout) -> LDS -> A-layout f16. Same-wave DS ops are in-order.
#pragma unroll
    for (int i = 0; i < 8; ++i) {
      const int r = half * 8 + i;
      pb[r * 36 + l16]      = t0[i];
      pb[r * 36 + 16 + l16] = t1[i];
    }
    asm volatile("" ::: "memory");
    v16h ap;
    {
      const float4* pp = (const float4*)(pb + l16 * 36 + half * 16);
      float4 p0 = pp[0], p1 = pp[1], p2 = pp[2], p3 = pp[3];
      ap[0]=(_Float16)p0.x;  ap[1]=(_Float16)p0.y;  ap[2]=(_Float16)p0.z;  ap[3]=(_Float16)p0.w;
      ap[4]=(_Float16)p1.x;  ap[5]=(_Float16)p1.y;  ap[6]=(_Float16)p1.z;  ap[7]=(_Float16)p1.w;
      ap[8]=(_Float16)p2.x;  ap[9]=(_Float16)p2.y;  ap[10]=(_Float16)p2.z; ap[11]=(_Float16)p2.w;
      ap[12]=(_Float16)p3.x; ap[13]=(_Float16)p3.y; ap[14]=(_Float16)p3.z; ap[15]=(_Float16)p3.w;
    }
    asm volatile("" ::: "memory");

#pragma unroll
    for (int f = 0; f < 4; ++f)
      accf[f] = wmma_f16(ap, bv[f], accf[f]);
  }

#pragma unroll
  for (int i = 0; i < 8; ++i) lrow[i] = 1.0f / lrow[i];
#pragma unroll
  for (int f = 0; f < 4; ++f)
#pragma unroll
    for (int i = 0; i < 8; ++i)
      out[(size_t)(gq0 + half * 8 + i) * HEAD + f * 16 + l16] = accf[f][i] * lrow[i];
}

// ---------------------------------------------------------------------------
extern "C" void kernel_launch(void* const* d_in, const int* in_sizes, int n_in,
                              void* d_out, int out_size, void* d_ws, size_t ws_size,
                              hipStream_t stream) {
  (void)in_sizes; (void)n_in; (void)out_size; (void)ws_size;
  const float* X     = (const float*)d_in[0];
  const float* Wq    = (const float*)d_in[1];
  const float* Wk    = (const float*)d_in[2];
  const float* Wv    = (const float*)d_in[3];
  const int*   pmask = (const int*)d_in[4];
  float*       out   = (float*)d_out;

  _Float16* Qh = (_Float16*)d_ws;                       // (B*T, H) f16, pre-scaled
  _Float16* Kh = Qh + (size_t)NBATCH * TSEQ * HEAD;     // (B*T, H) f16
  _Float16* Vt = Kh + (size_t)NBATCH * TSEQ * HEAD;     // (B, H, T) f16
  _Float16* Wh = Vt + (size_t)NBATCH * TSEQ * HEAD;     // (192, 1024) f16
  // total workspace: 6 MB + 384 KB

  convert_w_kernel<<<dim3(192), dim3(256), 0, stream>>>(Wq, Wk, Wv, Wh);

  const int tiles  = (NBATCH * TSEQ) / 16;              // 1024
  const int blocks = tiles / 8;                         // 128 blocks x 8 waves
  qkv_proj_kernel<<<dim3(blocks), dim3(256), 0, stream>>>(X, Wh, Qh, Kh, Vt);
  flash_attn_kernel<<<dim3(blocks), dim3(256), 0, stream>>>(Qh, Kh, Vt, pmask, out);
}
